// MultiHeadAttentionLayer_47330539602183
// MI455X (gfx1250) — matrix operand order repaired and checked by reading, not compiled, not run on hardware
//
#include <hip/hip_runtime.h>
#include <math.h>

// GraphTransformer attention layer for MI455X (gfx1250, wave32, WMMA).
//
// Sizes (fixed by the reference):
#define NN 50000      // nodes
#define EE 1600000    // edges
#define HH 8          // heads
#define DDIM 8        // dim per head  (HH*DDIM == 64 == IN)
// N and E are both multiples of 16, so every 16-row WMMA tile is full.
//   NN/16 = 3125 tiles, EE/16 = 100000 tiles (= 12500 blocks * 8 waves exactly)

typedef __attribute__((ext_vector_type(16))) _Float16 v16h;
typedef __attribute__((ext_vector_type(8)))  float    v8f;

// ---------------------------------------------------------------------------
// f32->f16 conversion into the WMMA A-operand lane layout.
// 16-bit A matrix 16x32 (ISA 7.12.2): lane L holds row M = L%16.
//   lanes 0-15 : halves 0..7 = K 0..7,   halves 8..15 = K 16..23
//   lanes 16-31: halves 0..7 = K 8..15,  halves 8..15 = K 24..31
// So per lane: two contiguous 8-float runs starting at ka and ka+16.
// ---------------------------------------------------------------------------
__device__ inline v16h cvt16(float4 f0, float4 f1, float4 f2, float4 f3) {
  v16h a;
  a[0]  = (_Float16)f0.x; a[1]  = (_Float16)f0.y; a[2]  = (_Float16)f0.z; a[3]  = (_Float16)f0.w;
  a[4]  = (_Float16)f1.x; a[5]  = (_Float16)f1.y; a[6]  = (_Float16)f1.z; a[7]  = (_Float16)f1.w;
  a[8]  = (_Float16)f2.x; a[9]  = (_Float16)f2.y; a[10] = (_Float16)f2.z; a[11] = (_Float16)f2.w;
  a[12] = (_Float16)f3.x; a[13] = (_Float16)f3.y; a[14] = (_Float16)f3.z; a[15] = (_Float16)f3.w;
  return a;
}

__device__ inline void load_A(const float* __restrict__ rowp, int lane, v16h& a0, v16h& a1) {
  const float4* rp = (const float4*)rowp;     // 16 float4 per 64-float row
  int ka = (lane < 16) ? 0 : 2;               // in float4 units (k=0 or 8)
  int kb = ka + 4;                            // k=16 or 24
  a0 = cvt16(rp[ka], rp[ka + 1], rp[kb], rp[kb + 1]);          // K 0..31
  a1 = cvt16(rp[ka + 8], rp[ka + 9], rp[kb + 8], rp[kb + 9]);  // K 32..63
}

// ---------------------------------------------------------------------------
// Pack the four 64x64 weight matrices (row-major [out,in], y = x @ W^T) into
// f16 WMMA B-operand layout. B is 32x16 (KxN); per the ISA B convention:
//   lane L holds column n = L%16; lanes 0-15 hold K 0..7,16..23 in halves 0..15,
//   lanes 16-31 hold K 8..15,24..31.  Two K-steps (K=0..31, 32..63) per column tile.
// Index: wp[(((m*4 + t)*2 + ks)*32 + lane)*16 + j]
// ---------------------------------------------------------------------------
__global__ void pack_weights_kernel(const float* __restrict__ Qw,
                                    const float* __restrict__ Kw,
                                    const float* __restrict__ Vw,
                                    const float* __restrict__ Ew,
                                    _Float16* __restrict__ wp) {
  int i = blockIdx.x * blockDim.x + threadIdx.x;
  if (i >= 4 * 4 * 2 * 32 * 16) return;
  int j  = i & 15;
  int L  = (i >> 4) & 31;
  int ks = (i >> 9) & 1;
  int t  = (i >> 10) & 3;
  int m  = i >> 12;
  const float* W = (m == 0) ? Qw : (m == 1) ? Kw : (m == 2) ? Vw : Ew;
  int n = t * 16 + (L & 15);
  int k = ks * 32 + ((j & 8) << 1) + ((L >> 4) << 3) + (j & 7);
  wp[i] = (_Float16)W[n * 64 + k];
}

// Zero the wV accumulator (h_out region of d_out) and z normalizers.
__global__ void zero_kernel(float* __restrict__ hout, float* __restrict__ z) {
  int i = blockIdx.x * blockDim.x + threadIdx.x;
  if (i < NN * 64) hout[i] = 0.0f;
  else if (i < NN * 64 + NN * 8) z[i - NN * 64] = 0.0f;
}

// ---------------------------------------------------------------------------
// Node projections: Q,K,V = h @ W^T + b.  One 16-row tile per wave,
// 3 matrices x 4 column tiles x 2 K-steps = 24 v_wmma per tile.
// ---------------------------------------------------------------------------
__global__ void node_proj_kernel(const float* __restrict__ h,
                                 const _Float16* __restrict__ wp,
                                 const float* __restrict__ Qb,
                                 const float* __restrict__ Kb,
                                 const float* __restrict__ Vb,
                                 float* __restrict__ Qo,
                                 float* __restrict__ Ko,
                                 float* __restrict__ Vo) {
  int wave = threadIdx.x >> 5;
  int lane = threadIdx.x & 31;
  int tile = blockIdx.x * 8 + wave;
  if (tile >= NN / 16) return;
  int row0 = tile * 16;

  v16h a0, a1;
  load_A(h + (long)(row0 + (lane & 15)) * 64, lane, a0, a1);

  int n_lo  = lane & 15;
  int mbase = (lane >> 4) << 3;   // C/D layout: lanes 0-15 -> M 0..7, 16-31 -> M 8..15

  #pragma unroll
  for (int m = 0; m < 3; ++m) {
    const float* bias = (m == 0) ? Qb : (m == 1) ? Kb : Vb;
    float* out        = (m == 0) ? Qo : (m == 1) ? Ko : Vo;
    #pragma unroll
    for (int t = 0; t < 4; ++t) {
      v8f c = {};
      v16h b0 = *(const v16h*)(wp + (size_t)(((m * 4 + t) * 2 + 0) * 32 + lane) * 16);
      v16h b1 = *(const v16h*)(wp + (size_t)(((m * 4 + t) * 2 + 1) * 32 + lane) * 16);
      c = __builtin_amdgcn_wmma_f32_16x16x32_f16(false, a0, false, b0, (short)0, c, false, false);
      c = __builtin_amdgcn_wmma_f32_16x16x32_f16(false, a1, false, b1, (short)0, c, false, false);
      int n = t * 16 + n_lo;
      float bv = bias[n];
      #pragma unroll
      for (int r = 0; r < 8; ++r)
        out[(long)(row0 + mbase + r) * 64 + n] = c[r] + bv;
    }
  }
}

// ---------------------------------------------------------------------------
// Fused edge kernel. Per wave: one 16-edge tile.
//  1) pe = e @ Ew^T + Eb via WMMA, kept in LDS (never hits HBM)
//  2) score = K[src]*Q[dst]/sqrt(8) * pe  -> e_out (the 410 MB stream)
//  3) s = exp(clip(sum_d score)); scatter V[src]*s and s with f32 L2 atomics
// Lane mapping for phase 2/3: lane -> edge (lane&15), 4 heads per lane.
// ---------------------------------------------------------------------------
__global__ void edge_kernel(const float* __restrict__ e,
                            const int* __restrict__ src,
                            const int* __restrict__ dst,
                            const _Float16* __restrict__ wp,
                            const float* __restrict__ Eb,
                            const float* __restrict__ Qf,
                            const float* __restrict__ Kf,
                            const float* __restrict__ Vf,
                            float* __restrict__ hout,
                            float* __restrict__ z,
                            float* __restrict__ eout) {
  __shared__ float lds[8 * 16 * 64];          // 32 KB: 8 waves x 16 edges x 64
  int wave = threadIdx.x >> 5;
  int lane = threadIdx.x & 31;
  long tile = (long)blockIdx.x * 8 + wave;    // EE/16 tiles exactly
  long e0 = tile * 16;
  float* myl = lds + wave * (16 * 64);

  int elocal = lane & 15;
  long edge = e0 + elocal;
  int sN = src[edge];
  int dN = dst[edge];
  // Warm the L2/L0 path for the gathers while the WMMAs run.
  __builtin_prefetch(Kf + (long)sN * 64, 0, 0);
  __builtin_prefetch(Qf + (long)dN * 64, 0, 0);
  __builtin_prefetch(Vf + (long)sN * 64, 0, 0);

  v16h a0, a1;
  load_A(e + edge * 64, lane, a0, a1);

  int n_lo  = lane & 15;
  int mbase = (lane >> 4) << 3;

  #pragma unroll
  for (int t = 0; t < 4; ++t) {
    v8f c = {};
    v16h b0 = *(const v16h*)(wp + (size_t)(((3 * 4 + t) * 2 + 0) * 32 + lane) * 16);
    v16h b1 = *(const v16h*)(wp + (size_t)(((3 * 4 + t) * 2 + 1) * 32 + lane) * 16);
    c = __builtin_amdgcn_wmma_f32_16x16x32_f16(false, a0, false, b0, (short)0, c, false, false);
    c = __builtin_amdgcn_wmma_f32_16x16x32_f16(false, a1, false, b1, (short)0, c, false, false);
    int n = t * 16 + n_lo;
    float bv = Eb[n];
    #pragma unroll
    for (int r = 0; r < 8; ++r)
      myl[(mbase + r) * 64 + n] = c[r] + bv;   // pe tile -> LDS
  }
  __syncthreads();   // all waves take identical path (E/16 divisible by 8)

  const float inv_sqrt_d = 0.35355339059327373f;  // 1/sqrt(8)
  #pragma unroll
  for (int hh = 0; hh < 4; ++hh) {
    int hd = (lane >> 4) + 2 * hh;               // heads 0..7 across half-waves
    const float4* Kp = (const float4*)(Kf + (long)sN * 64 + hd * 8);
    const float4* Qp = (const float4*)(Qf + (long)dN * 64 + hd * 8);
    const float4* Pp = (const float4*)(myl + elocal * 64 + hd * 8);
    float4 k0 = Kp[0], k1 = Kp[1];
    float4 q0 = Qp[0], q1 = Qp[1];
    float4 p0 = Pp[0], p1 = Pp[1];

    float s0 = k0.x * q0.x * inv_sqrt_d * p0.x;
    float s1 = k0.y * q0.y * inv_sqrt_d * p0.y;
    float s2 = k0.z * q0.z * inv_sqrt_d * p0.z;
    float s3 = k0.w * q0.w * inv_sqrt_d * p0.w;
    float s4 = k1.x * q1.x * inv_sqrt_d * p1.x;
    float s5 = k1.y * q1.y * inv_sqrt_d * p1.y;
    float s6 = k1.z * q1.z * inv_sqrt_d * p1.z;
    float s7 = k1.w * q1.w * inv_sqrt_d * p1.w;

    float4* Eo = (float4*)(eout + edge * 64 + hd * 8);
    Eo[0] = make_float4(s0, s1, s2, s3);
    Eo[1] = make_float4(s4, s5, s6, s7);

    float ssum = ((s0 + s1) + (s2 + s3)) + ((s4 + s5) + (s6 + s7));
    ssum = fminf(5.0f, fmaxf(-5.0f, ssum));
    float sexp = __expf(ssum);

    const float4* Vp = (const float4*)(Vf + (long)sN * 64 + hd * 8);
    float4 v0 = Vp[0], v1 = Vp[1];
    float* ho = hout + (long)dN * 64 + hd * 8;
    atomicAdd(ho + 0, v0.x * sexp);
    atomicAdd(ho + 1, v0.y * sexp);
    atomicAdd(ho + 2, v0.z * sexp);
    atomicAdd(ho + 3, v0.w * sexp);
    atomicAdd(ho + 4, v1.x * sexp);
    atomicAdd(ho + 5, v1.y * sexp);
    atomicAdd(ho + 6, v1.z * sexp);
    atomicAdd(ho + 7, v1.w * sexp);
    atomicAdd(z + (long)dN * 8 + hd, sexp);
  }
}

__global__ void norm_kernel(float* __restrict__ hout, const float* __restrict__ z) {
  int i = blockIdx.x * blockDim.x + threadIdx.x;
  if (i >= NN * 64) return;
  int node = i >> 6;
  int hd = (i & 63) >> 3;
  hout[i] = hout[i] / (z[node * 8 + hd] + 1e-6f);
}

// ---------------------------------------------------------------------------
extern "C" void kernel_launch(void* const* d_in, const int* in_sizes, int n_in,
                              void* d_out, int out_size, void* d_ws, size_t ws_size,
                              hipStream_t stream) {
  const float* h  = (const float*)d_in[0];
  const float* e  = (const float*)d_in[1];
  const int*   sr = (const int*)d_in[2];
  const int*   ds = (const int*)d_in[3];
  const float* Qw = (const float*)d_in[4];
  const float* Qb = (const float*)d_in[5];
  const float* Kw = (const float*)d_in[6];
  const float* Kb = (const float*)d_in[7];
  const float* Vw = (const float*)d_in[8];
  const float* Vb = (const float*)d_in[9];
  const float* Ew = (const float*)d_in[10];
  const float* Eb = (const float*)d_in[11];

  // Workspace: packed f16 weights (32 KB), then Q,K,V (f32 N x 64 each), z (N x 8).
  char* ws = (char*)d_ws;
  _Float16* wp = (_Float16*)ws;
  float* Qf = (float*)(ws + 32768);
  float* Kf = Qf + (size_t)NN * 64;
  float* Vf = Kf + (size_t)NN * 64;
  float* zf = Vf + (size_t)NN * 64;

  float* hout = (float*)d_out;                 // [N,H,D] accum then normalized
  float* eout = hout + (size_t)NN * 64;        // [E,H,D]

  pack_weights_kernel<<<64, 256, 0, stream>>>(Qw, Kw, Vw, Ew, wp);
  zero_kernel<<<(NN * 64 + NN * 8 + 255) / 256, 256, 0, stream>>>(hout, zf);
  node_proj_kernel<<<(NN / 16 + 7) / 8, 256, 0, stream>>>(h, wp, Qb, Kb, Vb, Qf, Kf, Vf);
  edge_kernel<<<EE / 16 / 8, 256, 0, stream>>>(e, sr, ds, wp, Eb, Qf, Kf, Vf, hout, zf, eout);
  norm_kernel<<<(NN * 64 + 255) / 256, 256, 0, stream>>>(hout, zf);

  (void)in_sizes; (void)n_in; (void)out_size; (void)ws_size;
}
